// MultiHeadAttention_88502096101602
// MI455X (gfx1250) — compile-verified, matrix operand order checked by reading, not run on hardware
//
#include <hip/hip_runtime.h>
#include <hip/hip_bf16.h>

// ---------------------------------------------------------------------------
// MultiHeadAttention forward for MI455X (gfx1250), bf16 WMMA + f32 accumulate,
// with Tensor-Data-Mover (TDM) staging of K/V tiles into LDS.
// B=4, S=2048, D=1024, H=16, HD=64.
// d_out = [ output (B*S*D f32) | attn_weights (B*H*S*S f32) ]
// ---------------------------------------------------------------------------

typedef __attribute__((ext_vector_type(8)))  float  v8f;
typedef __attribute__((ext_vector_type(8)))  __bf16 v8bf;
typedef __attribute__((ext_vector_type(16))) __bf16 v16bf;
typedef __attribute__((ext_vector_type(4)))  unsigned int u32x4;
typedef __attribute__((ext_vector_type(4)))  int i32x4;
typedef __attribute__((ext_vector_type(8)))  int i32x8;

static constexpr int Bsz = 4, S = 2048, D = 1024, H = 16, HD = 64;

#define WMMA_BF16(a, b, c) \
  __builtin_amdgcn_wmma_f32_16x16x32_bf16(false, (a), false, (b), (short)0, (c), false, false)

__device__ __forceinline__ v16bf cat8(v8bf lo, v8bf hi) {
  return __builtin_shufflevector(lo, hi, 0, 1, 2, 3, 4, 5, 6, 7,
                                         8, 9, 10, 11, 12, 13, 14, 15);
}

// ---------------------------------------------------------------------------
// TDM: async 2D bf16 tile load Global -> LDS.  D# per CDNA5 ISA:
//  group0: count=1 | lds_addr | global_addr[56:0] | type=2
//  group1: data_size=2B; tensor_dim0=tile_w; tensor_dim1=tile_h;
//          tile_dim0=tile_w; tile_dim1=tile_h; tensor_dim0_stride=row_stride
//  groups 2/3: zero (2D tensor)
// ---------------------------------------------------------------------------
__device__ __forceinline__ void tdm_load_tile_2d(unsigned lds_off, const __bf16* gptr,
                                                 unsigned tile_w, unsigned tile_h,
                                                 unsigned row_stride) {
  unsigned long long ga = (unsigned long long)(const void*)gptr;
  u32x4 g0 = { 1u,                                             // count=1, user mode
               lds_off,                                        // LDS byte address
               (unsigned)(ga & 0xffffffffu),                   // global addr lo
               (unsigned)((ga >> 32) & 0x01ffffffu) | 0x80000000u };  // hi + type=2
  i32x8 g1 = { (int)(1u << 16),                                // data_size=1 -> 2 bytes
               (int)(tile_w << 16),                            // tensor_dim0[15:0] @ [63:48]
               (int)((tile_h << 16) | (tile_w >> 16)),         // dim0 hi | tensor_dim1 lo
               (int)(tile_w << 16),                            // tile_dim0 @ [127:112]
               (int)tile_h,                                    // tile_dim1 @ [143:128]
               (int)row_stride,                                // tensor_dim0_stride lo32
               0, 0 };
  i32x4 z4 = {0, 0, 0, 0};
#if defined(__clang_major__) && (__clang_major__ >= 23)
  i32x8 z8 = {0, 0, 0, 0, 0, 0, 0, 0};
  __builtin_amdgcn_tensor_load_to_lds(g0, g1, z4, z4, z8, 0);
#else
  __builtin_amdgcn_tensor_load_to_lds(g0, g1, z4, z4, 0);
#endif
}

// ---------------------------------------------------------------------------
// Projection GEMM: Y[M,N] = X[M,K] @ W[N,K]^T + bias,  M=8192, N=K=1024.
// IN_BF16: X is bf16 (else f32, converted on the fly).
// OUT_F32: write f32 row-major (final projection into d_out).
// TRANS_V: write bf16 transposed per head: Vt[((b*H+h)*HD+d)*S + s].
// ---------------------------------------------------------------------------
template <bool IN_BF16, bool OUT_F32, bool TRANS_V>
__global__ __launch_bounds__(128) void proj_kernel(const void* __restrict__ Xv,
                                                   const float* __restrict__ W,
                                                   const float* __restrict__ bias,
                                                   void* __restrict__ Yv) {
  constexpr int K = 1024, N = 1024;
  __shared__ alignas(16) __bf16 As[64][40];
  __shared__ alignas(16) __bf16 Bs[64][40];

  const int tid  = threadIdx.x;
  const int lane = tid & 31;
  const int w    = tid >> 5;
  const int Nb   = blockIdx.x * 64;
  const int Mb   = blockIdx.y * 64;
  const int mo   = (w >> 1) * 32;
  const int no   = (w & 1) * 32;
  const int r16  = lane & 15;
  const int half = lane >> 4;
  const int kb   = half * 8;

  const float*  Xf = (const float*)Xv;
  const __bf16* Xb = (const __bf16*)Xv;

  v8f acc[2][2] = {};

  for (int k0 = 0; k0 < K; k0 += 32) {
    if (IN_BF16) {
#pragma unroll
      for (int it = 0; it < 2; ++it) {
        int slot = tid + it * 128;
        int row = slot >> 2;
        int c8  = (slot & 3) * 8;
        *(v8bf*)&As[row][c8] = *(const v8bf*)(Xb + (size_t)(Mb + row) * K + k0 + c8);
      }
    } else {
#pragma unroll
      for (int it = 0; it < 4; ++it) {
        int slot = tid + it * 128;
        int row = slot >> 3;
        int c4  = (slot & 7) * 4;
        float4 f = *(const float4*)(Xf + (size_t)(Mb + row) * K + k0 + c4);
        As[row][c4 + 0] = (__bf16)f.x;
        As[row][c4 + 1] = (__bf16)f.y;
        As[row][c4 + 2] = (__bf16)f.z;
        As[row][c4 + 3] = (__bf16)f.w;
      }
    }
#pragma unroll
    for (int it = 0; it < 4; ++it) {
      int slot = tid + it * 128;
      int row = slot >> 3;
      int c4  = (slot & 7) * 4;
      float4 f = *(const float4*)(W + (size_t)(Nb + row) * K + k0 + c4);
      Bs[row][c4 + 0] = (__bf16)f.x;
      Bs[row][c4 + 1] = (__bf16)f.y;
      Bs[row][c4 + 2] = (__bf16)f.z;
      Bs[row][c4 + 3] = (__bf16)f.w;
    }
    if (k0 + 32 < K) {  // CDNA5 global_prefetch_b8 of next K tiles
      __builtin_prefetch((const char*)Xv +
                             ((size_t)(Mb + (tid >> 1)) * K + k0 + 32) * (IN_BF16 ? 2 : 4),
                         0, 1);
      __builtin_prefetch((const char*)W + ((size_t)(Nb + (tid >> 1)) * K + k0 + 32) * 4, 0, 1);
    }
    __syncthreads();

    v16bf a0 = cat8(*(const v8bf*)&As[mo + r16][kb],      *(const v8bf*)&As[mo + r16][kb + 16]);
    v16bf a1 = cat8(*(const v8bf*)&As[mo + 16 + r16][kb], *(const v8bf*)&As[mo + 16 + r16][kb + 16]);
    v16bf b0 = cat8(*(const v8bf*)&Bs[no + r16][kb],      *(const v8bf*)&Bs[no + r16][kb + 16]);
    v16bf b1 = cat8(*(const v8bf*)&Bs[no + 16 + r16][kb], *(const v8bf*)&Bs[no + 16 + r16][kb + 16]);

    acc[0][0] = WMMA_BF16(a0, b0, acc[0][0]);
    acc[0][1] = WMMA_BF16(a0, b1, acc[0][1]);
    acc[1][0] = WMMA_BF16(a1, b0, acc[1][0]);
    acc[1][1] = WMMA_BF16(a1, b1, acc[1][1]);
    __syncthreads();
  }

  float*  Yf = (float*)Yv;
  __bf16* Yb = (__bf16*)Yv;
#pragma unroll
  for (int i = 0; i < 2; ++i) {
#pragma unroll
    for (int j = 0; j < 2; ++j) {
      int   ncol = Nb + no + j * 16 + r16;
      float bval = bias[ncol];
#pragma unroll
      for (int rr = 0; rr < 8; ++rr) {
        int   m   = Mb + mo + i * 16 + rr + 8 * half;
        float val = acc[i][j][rr] + bval;
        if (OUT_F32) {
          Yf[(size_t)m * N + ncol] = val;
        } else if (TRANS_V) {
          int bb = m / S, ss = m % S;
          int hh = ncol / HD, dd = ncol % HD;
          Yb[((size_t)((bb * H + hh) * HD + dd)) * S + ss] = (__bf16)val;
        } else {
          Yb[(size_t)m * N + ncol] = (__bf16)val;
        }
      }
    }
  }
}

// ---------------------------------------------------------------------------
// Fused attention. Block = 4 waves sharing (b, h); wave w owns q-rows
// [qt0*64 + w*16, +16). K tiles (32 keys x 64d) and V tiles (64d x 32s) are
// TDM-staged into double-buffered LDS shared by all 4 waves.
// Dynamic LDS layout (base offset 0):
//   [    0, 8192)  Ks[2][32][64] bf16
//   [ 8192,16384)  Vs[2][64][32] bf16
//   [16384,20480)  Ps[4][16][32] bf16 (per-wave P staging)
// ---------------------------------------------------------------------------
__global__ __launch_bounds__(128) void attn_kernel(const __bf16* __restrict__ Qb,
                                                   const __bf16* __restrict__ Kb,
                                                   const __bf16* __restrict__ Vt,
                                                   const int* __restrict__ mask,
                                                   float* __restrict__ attnw,
                                                   __bf16* __restrict__ AO) {
  extern __shared__ __align__(16) char smem[];
  __bf16* Ks = (__bf16*)smem;                      // 2 x [32][64]
  __bf16* Vs = (__bf16*)(smem + 8192);             // 2 x [64][32]

  const int tid  = threadIdx.x;
  const int lane = tid & 31;
  const int w    = tid >> 5;
  __bf16* Ps = (__bf16*)(smem + 16384) + w * (16 * 32);

  const int qt0 = blockIdx.x & 31;                 // S/64 tiles of 64 rows
  const int h   = (blockIdx.x >> 5) & 15;
  const int b   = blockIdx.x >> 9;
  const int qbase = qt0 * 64 + w * 16;
  const int r16  = lane & 15;
  const int half = lane >> 4;
  const int kb   = half * 8;

  const __bf16* Qh = Qb + (size_t)b * S * D + (size_t)h * HD;
  const __bf16* Kh = Kb + (size_t)b * S * D + (size_t)h * HD;
  const __bf16* Vh = Vt + (size_t)(b * H + h) * HD * S;
  const int*    mk = mask + (size_t)b * S * S;
  float*        aw = attnw + (size_t)(b * H + h) * S * S;

  // Q fragments, resident for the whole kernel
  v16bf aq[2];
#pragma unroll
  for (int c = 0; c < 2; ++c) {
    const __bf16* p = Qh + (size_t)(qbase + r16) * D + c * 32 + kb;
    aq[c] = cat8(*(const v8bf*)p, *(const v8bf*)(p + 16));
  }

  auto issueK = [&](int jj, int buf) {
    tdm_load_tile_2d((unsigned)(buf * 4096), Kh + (size_t)(jj * 32) * D, 64u, 32u, (unsigned)D);
  };
  auto issueV = [&](int jj, int buf) {
    tdm_load_tile_2d((unsigned)(8192 + buf * 4096), Vh + (size_t)(jj * 32), 32u, 64u, (unsigned)S);
  };

  // scaled + masked 16x16 score tile; K from LDS tile kt32 ([32][64])
  auto score_lds = [&](const __bf16* kt32, int t, int jglob) -> v8f {
    v8f c = {};
#pragma unroll
    for (int ch = 0; ch < 2; ++ch) {
      const __bf16* p = kt32 + (size_t)(t * 16 + r16) * 64 + ch * 32 + kb;
      v16bf bf = cat8(*(const v8bf*)p, *(const v8bf*)(p + 16));
      c = WMMA_BF16(aq[ch], bf, c);
    }
#pragma unroll
    for (int rr = 0; rr < 8; ++rr) {
      int qrow = qbase + rr + 8 * half;
      int mv   = mk[(size_t)qrow * S + jglob * 16 + r16];
      c[rr] = (mv == 0) ? -1e9f : c[rr] * 0.125f;  // 1/sqrt(64)
    }
    return c;
  };

  // ---- pass 1: row max (K TDM double-buffered) ----
  float rm[8];
#pragma unroll
  for (int rr = 0; rr < 8; ++rr) rm[rr] = -3.0e38f;
  if (tid == 0) issueK(0, 0);
  for (int jj = 0; jj < S / 32; ++jj) {
    int buf = jj & 1;
    if (tid == 0) {
      if (jj < S / 32 - 1) {
        issueK(jj + 1, buf ^ 1);
        __builtin_amdgcn_s_wait_tensorcnt((short)1);
      } else {
        __builtin_amdgcn_s_wait_tensorcnt((short)0);
      }
    }
    __syncthreads();
    const __bf16* kt = Ks + buf * 2048;
#pragma unroll
    for (int t = 0; t < 2; ++t) {
      v8f c = score_lds(kt, t, jj * 2 + t);
#pragma unroll
      for (int rr = 0; rr < 8; ++rr) rm[rr] = fmaxf(rm[rr], c[rr]);
    }
    __syncthreads();
  }
#pragma unroll
  for (int rr = 0; rr < 8; ++rr) {
#pragma unroll
    for (int off = 1; off < 16; off <<= 1)
      rm[rr] = fmaxf(rm[rr], __shfl_xor(rm[rr], off, 32));
  }

  // ---- pass 2: exp + row sums + P@V (K and V TDM double-buffered) ----
  float rs[8] = {0.f, 0.f, 0.f, 0.f, 0.f, 0.f, 0.f, 0.f};
  v8f   o[4] = {};
  if (tid == 0) { issueK(0, 0); issueV(0, 0); }
  for (int jj = 0; jj < S / 32; ++jj) {
    int buf = jj & 1;
    if (tid == 0) {
      if (jj < S / 32 - 1) {
        issueK(jj + 1, buf ^ 1);
        issueV(jj + 1, buf ^ 1);
        __builtin_amdgcn_s_wait_tensorcnt((short)2);
      } else {
        __builtin_amdgcn_s_wait_tensorcnt((short)0);
      }
    }
    __syncthreads();
    const __bf16* kt = Ks + buf * 2048;
#pragma unroll
    for (int t = 0; t < 2; ++t) {
      v8f c = score_lds(kt, t, jj * 2 + t);
#pragma unroll
      for (int rr = 0; rr < 8; ++rr) {
        float p = __expf(c[rr] - rm[rr]);
        rs[rr] += p;
        Ps[(rr + 8 * half) * 32 + t * 16 + r16] = (__bf16)p;  // C -> row-major
      }
    }
    // same-wave LDS ops are in-order: safe to reload as A fragment
    v16bf ap = cat8(*(const v8bf*)&Ps[r16 * 32 + kb], *(const v8bf*)&Ps[r16 * 32 + kb + 16]);
    const __bf16* vt = Vs + buf * 2048;
#pragma unroll
    for (int n = 0; n < 4; ++n) {
      const __bf16* vp = vt + (size_t)(n * 16 + r16) * 32 + kb;
      v16bf bv = cat8(*(const v8bf*)vp, *(const v8bf*)(vp + 16));
      o[n] = WMMA_BF16(ap, bv, o[n]);
    }
    __syncthreads();
  }
  float inv[8];
#pragma unroll
  for (int rr = 0; rr < 8; ++rr) {
#pragma unroll
    for (int off = 1; off < 16; off <<= 1)
      rs[rr] += __shfl_xor(rs[rr], off, 32);
    inv[rr] = 1.0f / rs[rr];
  }

  // ---- pass 3: write normalized attention weights ----
  if (tid == 0) issueK(0, 0);
  for (int jj = 0; jj < S / 32; ++jj) {
    int buf = jj & 1;
    if (tid == 0) {
      if (jj < S / 32 - 1) {
        issueK(jj + 1, buf ^ 1);
        __builtin_amdgcn_s_wait_tensorcnt((short)1);
      } else {
        __builtin_amdgcn_s_wait_tensorcnt((short)0);
      }
    }
    __syncthreads();
    const __bf16* kt = Ks + buf * 2048;
#pragma unroll
    for (int t = 0; t < 2; ++t) {
      v8f c = score_lds(kt, t, jj * 2 + t);
#pragma unroll
      for (int rr = 0; rr < 8; ++rr) {
        int qrow = qbase + rr + 8 * half;
        aw[(size_t)qrow * S + (jj * 2 + t) * 16 + r16] = __expf(c[rr] - rm[rr]) * inv[rr];
      }
    }
    __syncthreads();
  }

  // ---- normalized attention output (merged heads, bf16 workspace) ----
#pragma unroll
  for (int n = 0; n < 4; ++n) {
#pragma unroll
    for (int rr = 0; rr < 8; ++rr) {
      int qrow = qbase + rr + 8 * half;
      AO[(size_t)(b * S + qrow) * D + h * HD + n * 16 + r16] = (__bf16)(o[n][rr] * inv[rr]);
    }
  }
}

// ---------------------------------------------------------------------------
extern "C" void kernel_launch(void* const* d_in, const int* in_sizes, int n_in,
                              void* d_out, int out_size, void* d_ws, size_t ws_size,
                              hipStream_t stream) {
  (void)in_sizes; (void)n_in; (void)out_size; (void)ws_size;

  const float* q    = (const float*)d_in[0];
  const float* k    = (const float*)d_in[1];
  const float* v    = (const float*)d_in[2];
  const int*   mask = (const int*)d_in[3];
  const float* Wq   = (const float*)d_in[4];
  const float* bq   = (const float*)d_in[5];
  const float* Wk   = (const float*)d_in[6];
  const float* bk   = (const float*)d_in[7];
  const float* Wv   = (const float*)d_in[8];
  const float* bv   = (const float*)d_in[9];
  const float* Wo   = (const float*)d_in[10];
  const float* bo   = (const float*)d_in[11];

  float* out   = (float*)d_out;
  float* attnw = out + (size_t)Bsz * S * D;

  char*        ws  = (char*)d_ws;
  const size_t seg = (size_t)Bsz * S * D * sizeof(__bf16);  // 16 MiB each
  __bf16* Qb = (__bf16*)(ws);
  __bf16* Kb = (__bf16*)(ws + seg);
  __bf16* Vt = (__bf16*)(ws + 2 * seg);
  __bf16* AO = (__bf16*)(ws + 3 * seg);

  dim3 pgrid(D / 64, (Bsz * S) / 64);  // 16 x 128
  dim3 pblk(128);
  proj_kernel<false, false, false><<<pgrid, pblk, 0, stream>>>(q, Wq, bq, Qb);
  proj_kernel<false, false, false><<<pgrid, pblk, 0, stream>>>(k, Wk, bk, Kb);
  proj_kernel<false, false, true ><<<pgrid, pblk, 0, stream>>>(v, Wv, bv, Vt);

  // B*H*(S/64) = 2048 blocks, 20 KiB dynamic LDS (K/V double buffers + P stage)
  attn_kernel<<<dim3(Bsz * H * (S / 64)), pblk, 20480, stream>>>(Qb, Kb, Vt, mask, attnw, AO);

  proj_kernel<true, true, false><<<pgrid, pblk, 0, stream>>>(AO, Wo, bo, out);
}
